// Level_14276471292451
// MI455X (gfx1250) — matrix operand order repaired and checked by reading, not compiled, not run on hardware
//
#include <hip/hip_runtime.h>

typedef __attribute__((ext_vector_type(16))) _Float16 v16h;
typedef __attribute__((ext_vector_type(8)))  _Float16 v8h;
typedef __attribute__((ext_vector_type(8)))  float    v8f;
typedef __attribute__((ext_vector_type(4)))  float    v4f;

#define Bb   16
#define NN   4096
#define TFD  512
#define DMD  1024
#define MROWS (Bb*NN)          // 65536 rows of the (B*N, DM) activation matrix
#define NCH  16                // scan chunks per sequence
#define CH   (NN/NCH)          // 256 steps per chunk

typedef __attribute__((address_space(3))) _Float16 lds_h;

// 32-bit LDS byte offset of a generic pointer into a __shared__ array
__device__ __forceinline__ unsigned lds_off(const void* p) {
    return (unsigned)(unsigned long long)(lds_h*)p;
}

// async DMA: 16 bytes global -> LDS, tracked by ASYNCcnt (CDNA5 §15.18 op 98)
__device__ __forceinline__ void async_g2lds_b128(unsigned lds_addr, const void* gaddr) {
    asm volatile("global_load_async_to_lds_b128 %0, %1, off"
                 :: "v"(lds_addr), "v"((unsigned long long)gaddr)
                 : "memory");
}

__device__ __forceinline__ void wait_asynccnt0() {
    asm volatile("s_wait_asynccnt 0x0" ::: "memory");
}

// ---------------------------------------------------------------------------
// Kernel 0: pre-convert w_seasonal (TF x DM fp32) into f16 hi/lo planes (2 MB)
// ---------------------------------------------------------------------------
__global__ void k_convert_w(const float* __restrict__ w,
                            _Float16* __restrict__ whi,
                            _Float16* __restrict__ wlo, int n) {
    int i = blockIdx.x * blockDim.x + threadIdx.x;
    if (i < n) {
        float f = w[i];
        _Float16 h = (_Float16)f;
        whi[i] = h;
        wlo[i] = (_Float16)(f - (float)h);
    }
}

// ---------------------------------------------------------------------------
// Kernel 1: z = 2*x - (latent_seasonal @ w_seasonal^T + b_seasonal) -> d_out
// WMMA f16 hi/lo split, fp32 accumulate. Block tile 128(M) x 64(F), 8 waves.
// B tiles staged via GLOBAL_LOAD_ASYNC_TO_LDS_B128 (ASYNCcnt), A tiles
// converted fp32->f16 hi/lo in-register while the async DMA runs.
// ---------------------------------------------------------------------------
__global__ __launch_bounds__(256) void k_gemm_z(
    const float*    __restrict__ x,
    const float*    __restrict__ latent,
    const _Float16* __restrict__ whi,
    const _Float16* __restrict__ wlo,
    const float*    __restrict__ bias,
    float*          __restrict__ out)
{
    __shared__ __attribute__((aligned(16))) _Float16 Ahi[128][40]; // +8 pad halves
    __shared__ __attribute__((aligned(16))) _Float16 Alo[128][40];
    __shared__ __attribute__((aligned(16))) _Float16 Bhi[64][40];
    __shared__ __attribute__((aligned(16))) _Float16 Blo[64][40];

    const int tid  = threadIdx.x;
    const int lane = tid & 31;
    const int wid  = tid >> 5;
    const int wm   = wid & 3;   // wave row   (0..3) -> 32 M-rows each
    const int wf   = wid >> 2;  // wave col   (0..1) -> 32 F-cols each
    const int row0 = blockIdx.x * 128;
    const int f0   = blockIdx.y * 64;

    // per-thread B-tile slot: 16B per plane per K-step
    const int br  = tid >> 2;            // 0..63 (f row of tile)
    const int bc8 = (tid & 3) * 8;       // k offset 0,8,16,24
    const unsigned bhi_lds = lds_off(&Bhi[br][bc8]);
    const unsigned blo_lds = lds_off(&Blo[br][bc8]);

    v8f zero = {};
    v8f acc[2][2];
    #pragma unroll
    for (int mi = 0; mi < 2; ++mi)
        #pragma unroll
        for (int fi = 0; fi < 2; ++fi) acc[mi][fi] = zero;

    for (int k0 = 0; k0 < DMD; k0 += 32) {
        // ---- kick off async B-tile DMA (64 x 32 halves hi + lo) ----
        {
            size_t gi = (size_t)(f0 + br) * DMD + k0 + bc8;
            async_g2lds_b128(bhi_lds, &whi[gi]);
            async_g2lds_b128(blo_lds, &wlo[gi]);
        }
        // ---- stage A tile (128 x 32 fp32 -> f16 hi/lo in LDS) while DMA runs
        #pragma unroll
        for (int it = 0; it < 4; ++it) {
            int idx = tid + it * 256;        // 0..1023 float4 slots
            int r   = idx >> 3;              // row 0..127
            int c4  = (idx & 7) * 4;         // k offset 0..28
            const float* gp = &latent[(size_t)(row0 + r) * DMD + k0 + c4];
            __builtin_prefetch(gp + 32, 0, 3); // next K tile (speculative; OOB dropped)
            v4f v = *(const v4f*)gp;
            #pragma unroll
            for (int e = 0; e < 4; ++e) {
                float f = v[e];
                _Float16 h = (_Float16)f;
                Ahi[r][c4 + e] = h;
                Alo[r][c4 + e] = (_Float16)(f - (float)h);
            }
        }
        wait_asynccnt0();                    // B tile landed in LDS
        __syncthreads();

        // ---- build fragments (ISA 16-bit layouts) ----
        // A (16x32): lane m<16 holds K[0..7],K[16..23]; lane m+16 holds K[8..15],K[24..31]
        // B (32x16): lane n<16 holds K[0..7],K[16..23]; lane n+16 holds K[8..15],K[24..31]
        const int mr = lane & 15;
        const int ka = (lane >> 4) * 8;
        const int kb = (lane >> 4) * 8;
        v16h afh[2], afl[2], bfh[2], bfl[2];
        #pragma unroll
        for (int mi = 0; mi < 2; ++mi) {
            int r = wm * 32 + mi * 16 + mr;
            v8h p0 = *(const v8h*)&Ahi[r][ka];
            v8h p1 = *(const v8h*)&Ahi[r][ka + 16];
            v8h q0 = *(const v8h*)&Alo[r][ka];
            v8h q1 = *(const v8h*)&Alo[r][ka + 16];
            #pragma unroll
            for (int e = 0; e < 8; ++e) {
                afh[mi][e] = p0[e]; afh[mi][8 + e] = p1[e];
                afl[mi][e] = q0[e]; afl[mi][8 + e] = q1[e];
            }
        }
        #pragma unroll
        for (int fi = 0; fi < 2; ++fi) {
            int r = wf * 32 + fi * 16 + mr;  // output col == Btile row (w row)
            v8h p0 = *(const v8h*)&Bhi[r][kb];
            v8h p1 = *(const v8h*)&Bhi[r][kb + 16];
            v8h q0 = *(const v8h*)&Blo[r][kb];
            v8h q1 = *(const v8h*)&Blo[r][kb + 16];
            #pragma unroll
            for (int e = 0; e < 8; ++e) {
                bfh[fi][e] = p0[e]; bfh[fi][8 + e] = p1[e];
                bfl[fi][e] = q0[e]; bfl[fi][8 + e] = q1[e];
            }
        }

        // ---- 12 WMMAs per K-step: hi*hi + hi*lo + lo*hi (fp32 acc) ----
        #pragma unroll
        for (int mi = 0; mi < 2; ++mi)
            #pragma unroll
            for (int fi = 0; fi < 2; ++fi) {
                acc[mi][fi] = __builtin_amdgcn_wmma_f32_16x16x32_f16(
                    false, afh[mi], false, bfh[fi], (short)0, acc[mi][fi], false, false);
                acc[mi][fi] = __builtin_amdgcn_wmma_f32_16x16x32_f16(
                    false, afh[mi], false, bfl[fi], (short)0, acc[mi][fi], false, false);
                acc[mi][fi] = __builtin_amdgcn_wmma_f32_16x16x32_f16(
                    false, afl[mi], false, bfh[fi], (short)0, acc[mi][fi], false, false);
            }
        __syncthreads();
    }

    // ---- epilogue: z = 2x - (gemm + bias); C/D layout: vgpr i -> M=i(+8), lane&15 -> N
    const int mr  = lane & 15;
    const int mhi = (lane >> 4) * 8;
    #pragma unroll
    for (int mi = 0; mi < 2; ++mi)
        #pragma unroll
        for (int fi = 0; fi < 2; ++fi) {
            int fcol = f0 + wf * 32 + fi * 16 + mr;
            float bv = bias[fcol];
            #pragma unroll
            for (int i = 0; i < 8; ++i) {
                int grow = row0 + wm * 32 + mi * 16 + mhi + i;
                size_t gi = (size_t)grow * TFD + fcol;
                out[gi] = 2.0f * x[gi] - (acc[mi][fi][i] + bv);
            }
        }
}

// ---------------------------------------------------------------------------
// Kernel 2: per-chunk local EMA (in place on d_out), store chunk-end states
// ---------------------------------------------------------------------------
__global__ void k_scan_local(float* __restrict__ out, float* __restrict__ S,
                             const float* __restrict__ alpha) {
    int t = blockIdx.x * blockDim.x + threadIdx.x;   // B*NCH*TF threads
    int f = t % TFD;
    int c = (t / TFD) % NCH;
    int b = t / (TFD * NCH);
    float a   = 1.0f / (1.0f + __expf(-alpha[0]));
    float lam = 1.0f - a;
    size_t base = ((size_t)b * NN + (size_t)c * CH) * TFD + f;
    float s = 0.0f;
    for (int j = 0; j < CH; ++j) {
        size_t gi = base + (size_t)j * TFD;
        s = lam * s + a * out[gi];
        out[gi] = s;
    }
    S[((size_t)b * NCH + c) * TFD + f] = s;
}

// ---------------------------------------------------------------------------
// Kernel 3: combine chunk-end states into carries:  T[c] = E[c] + lam^CH * T[c-1]
// ---------------------------------------------------------------------------
__global__ void k_carries(const float* __restrict__ S, float* __restrict__ Carry,
                          const float* __restrict__ alpha) {
    int t = blockIdx.x * blockDim.x + threadIdx.x;   // B*TF threads
    int f = t % TFD;
    int b = t / TFD;
    float a    = 1.0f / (1.0f + __expf(-alpha[0]));
    float lam  = 1.0f - a;
    float lamC = __powf(lam, (float)CH);
    float T = S[((size_t)b * NCH) * TFD + f];
    for (int c = 1; c < NCH; ++c) {
        Carry[((size_t)b * NCH + c) * TFD + f] = T;  // carry INTO chunk c
        T = S[((size_t)b * NCH + c) * TFD + f] + lamC * T;
    }
}

// ---------------------------------------------------------------------------
// Kernel 4: apply carries: out[b, c*CH+j, f] += lam^(j+1) * carry  (c >= 1)
// ---------------------------------------------------------------------------
__global__ void k_apply(float* __restrict__ out, const float* __restrict__ Carry,
                        const float* __restrict__ alpha) {
    int t = blockIdx.x * blockDim.x + threadIdx.x;   // B*(NCH-1)*CH*TF threads
    int f = t % TFD;
    int j = (t / TFD) % CH;
    int c = 1 + (t / (TFD * CH)) % (NCH - 1);
    int b = t / (TFD * CH * (NCH - 1));
    float a   = 1.0f / (1.0f + __expf(-alpha[0]));
    float lam = 1.0f - a;
    float T = Carry[((size_t)b * NCH + c) * TFD + f];
    float w = __powf(lam, (float)(j + 1));
    size_t gi = ((size_t)b * NN + (size_t)c * CH + j) * TFD + f;
    out[gi] += w * T;
}

// ---------------------------------------------------------------------------
// Host launch. Inputs: 0=x, 1=latent_growth(unused), 2=latent_seasonal,
// 3=alpha, 4=w_growth(unused), 5=b_growth(unused), 6=w_seasonal, 7=b_seasonal
// ws layout: [0,1MB) Whi | [1,2MB) Wlo | [2,2.5MB) S | [2.5,3MB) Carry
// ---------------------------------------------------------------------------
extern "C" void kernel_launch(void* const* d_in, const int* in_sizes, int n_in,
                              void* d_out, int out_size, void* d_ws, size_t ws_size,
                              hipStream_t stream) {
    const float* x      = (const float*)d_in[0];
    const float* latent = (const float*)d_in[2];
    const float* alpha  = (const float*)d_in[3];
    const float* wseas  = (const float*)d_in[6];
    const float* bseas  = (const float*)d_in[7];
    float* out = (float*)d_out;

    char* ws = (char*)d_ws;
    _Float16* whi = (_Float16*)ws;
    _Float16* wlo = (_Float16*)(ws + (size_t)TFD * DMD * 2);
    float* S      = (float*)(ws + (size_t)TFD * DMD * 4);
    float* Carry  = (float*)(ws + (size_t)TFD * DMD * 4 + (size_t)Bb * NCH * TFD * 4);

    // 0) weights -> f16 hi/lo
    k_convert_w<<<(TFD * DMD) / 256, 256, 0, stream>>>(wseas, whi, wlo, TFD * DMD);
    // 1) WMMA GEMM + fused z = 2x - seasonal  -> d_out
    k_gemm_z<<<dim3(MROWS / 128, TFD / 64), 256, 0, stream>>>(x, latent, whi, wlo, bseas, out);
    // 2) local chunk EMA (in place) + chunk-end states
    k_scan_local<<<(Bb * NCH * TFD) / 256, 256, 0, stream>>>(out, S, alpha);
    // 3) carry combine across chunks
    k_carries<<<(Bb * TFD) / 256, 256, 0, stream>>>(S, Carry, alpha);
    // 4) parallel carry application
    k_apply<<<(Bb * (NCH - 1) * CH * TFD) / 256, 256, 0, stream>>>(out, Carry, alpha);
}
